// MoELayer_33621003993991
// MI455X (gfx1250) — matrix operand order repaired and checked by reading, not compiled, not run on hardware
//
#include <hip/hip_runtime.h>
#include <math.h>

#define NTOK 4096
#define DDIM 1024
#define FDIM 4096
#define NEXP 8
#define TOTROWS (NTOK * 2)

#define BM 128
#define BN 128
#define BK 32
#define LDA 40
#define LDB 40

typedef __bf16 bf16;
typedef __attribute__((ext_vector_type(16))) __bf16 v16bf;
typedef __attribute__((ext_vector_type(2)))  __bf16 bf16x2;
typedef __attribute__((ext_vector_type(2)))  float  f32x2;
typedef __attribute__((ext_vector_type(8)))  float  v8f;
typedef __attribute__((ext_vector_type(8)))  unsigned int v8u;

static __device__ __forceinline__ v16bf as_v16bf(v8u x) {
    union { v8u u; v16bf b; } c; c.u = x; return c.b;
}
// One v_cvt_pk_bf16_f32: two f32 -> packed bf16x2 in a dword.
static __device__ __forceinline__ unsigned pack2(float lo, float hi) {
    f32x2 f = { lo, hi };
    bf16x2 h = __builtin_convertvector(f, bf16x2);
    union { bf16x2 h; unsigned u; } c; c.h = h;
    return c.u;
}

// Shared WMMA micro-kernel: one K=32 step on a 128x128 tile, wave does 32x64.
static __device__ __forceinline__ void wmma_step(
    const bf16* __restrict__ tA, const bf16* __restrict__ tB,
    v8f (&acc)[2][4], int wm, int wn, int am, int khalf)
{
    v8u aF[2];
#pragma unroll
    for (int sm = 0; sm < 2; ++sm) {
        int m = wm + sm * 16 + am;
#pragma unroll
        for (int j = 0; j < 8; ++j) {
            int kk = ((j < 4) ? 2 * j : 16 + 2 * (j - 4)) + khalf * 8;
            aF[sm][j] = *(const unsigned int*)&tA[m * LDA + kk];
        }
    }
#pragma unroll
    for (int sn = 0; sn < 4; ++sn) {
        v8u bF;
        int n = wn + sn * 16 + am;
#pragma unroll
        for (int j = 0; j < 8; ++j) {
            int kk = ((j < 4) ? 2 * j : 16 + 2 * (j - 4)) + khalf * 8;
            bF[j] = *(const unsigned int*)&tB[n * LDB + kk];
        }
        v16bf bb = as_v16bf(bF);
#pragma unroll
        for (int sm = 0; sm < 2; ++sm)
            acc[sm][sn] = __builtin_amdgcn_wmma_f32_16x16x32_bf16(
                false, as_v16bf(aF[sm]), false, bb, (short)0, acc[sm][sn],
                false, false);
    }
}

// ---------------- Router: one wave per token ----------------
__global__ __launch_bounds__(256) void moe_router(
    const float* __restrict__ x, const float* __restrict__ Wr,
    int* __restrict__ topk_idx, float* __restrict__ topk_w)
{
    int wave = blockIdx.x * (blockDim.x >> 5) + (threadIdx.x >> 5);
    int lane = threadIdx.x & 31;
    if (wave >= NTOK) return;
    const float* xr = x + (size_t)wave * DDIM;

    float acc[NEXP];
#pragma unroll
    for (int e = 0; e < NEXP; ++e) acc[e] = 0.f;
    for (int d = lane; d < DDIM; d += 32) {
        float xv = xr[d];
        const float* wr = Wr + d * NEXP;
#pragma unroll
        for (int e = 0; e < NEXP; ++e) acc[e] += xv * wr[e];
    }
#pragma unroll
    for (int e = 0; e < NEXP; ++e)
        for (int off = 16; off > 0; off >>= 1)
            acc[e] += __shfl_xor(acc[e], off, 32);

    if (lane == 0) {
        float m = acc[0];
        for (int e = 1; e < NEXP; ++e) m = fmaxf(m, acc[e]);
        float p[NEXP];
        for (int e = 0; e < NEXP; ++e) p[e] = __expf(acc[e] - m);
        int i0 = 0; float v0 = p[0];
        for (int e = 1; e < NEXP; ++e) if (p[e] > v0) { v0 = p[e]; i0 = e; }
        int i1 = (i0 == 0) ? 1 : 0; float v1 = p[i1];
        for (int e = 0; e < NEXP; ++e)
            if (e != i0 && p[e] > v1) { v1 = p[e]; i1 = e; }
        float s = v0 + v1;
        topk_idx[wave * 2 + 0] = i0;
        topk_idx[wave * 2 + 1] = i1;
        topk_w[wave * 2 + 0] = v0 / s;
        topk_w[wave * 2 + 1] = v1 / s;
    }
}

// ---------------- Compaction: per-expert token lists (1 block) ----------------
__global__ __launch_bounds__(256) void moe_compact(
    const int* __restrict__ topk_idx, const float* __restrict__ topk_w,
    int* __restrict__ etok, float* __restrict__ ewt, int* __restrict__ eoff)
{
    __shared__ int s_wsum[8];
    __shared__ int s_run;
    int tid = threadIdx.x;
    int lane = tid & 31, wid = tid >> 5;
    if (tid == 0) { s_run = 0; eoff[0] = 0; }
    __syncthreads();

    for (int e = 0; e < NEXP; ++e) {
        for (int base = 0; base < NTOK; base += 256) {
            int t = base + tid;
            int i0 = topk_idx[t * 2], i1 = topk_idx[t * 2 + 1];
            bool flag = (i0 == e) || (i1 == e);
            float w = (i0 == e) ? topk_w[t * 2] : topk_w[t * 2 + 1];
            unsigned int mlo = (unsigned int)__ballot(flag);
            int rank = __popc(mlo & ((1u << lane) - 1u));
            if (lane == 0) s_wsum[wid] = __popc(mlo);
            __syncthreads();
            int wbase = 0, btot = 0;
            for (int i = 0; i < 8; ++i) {
                if (i < wid) wbase += s_wsum[i];
                btot += s_wsum[i];
            }
            if (flag) {
                int pos = s_run + wbase + rank;
                etok[pos] = t;
                ewt[pos]  = w;
            }
            __syncthreads();
            if (tid == 0) s_run += btot;
            __syncthreads();
        }
        if (tid == 0) eoff[e + 1] = s_run;
        __syncthreads();
    }
}

__global__ void moe_zero(float* __restrict__ p, int n) {
    int i = blockIdx.x * blockDim.x + threadIdx.x;
    if (i < n) p[i] = 0.f;
}

// ---------------- FFN1: H = gelu(gather(x) @ W1[e] + b1[e]) ----------------
__global__ __launch_bounds__(256) void moe_ffn1(
    const float* __restrict__ x, const float* __restrict__ W1,
    const float* __restrict__ b1, const int* __restrict__ etok,
    const int* __restrict__ eoff, bf16* __restrict__ H)
{
    __shared__ bf16 lA[2][BM * LDA];
    __shared__ bf16 lB[2][BN * LDB];
    __shared__ int  s_tok[BM];

    const int e = blockIdx.y >> 5;
    const int tile_m = blockIdx.y & 31;
    const int off0 = eoff[e];
    const int cnt  = eoff[e + 1] - off0;
    if (tile_m * BM >= cnt) return;
    const int n_base = blockIdx.x * BN;

    const int tid = threadIdx.x;
    if (tid < BM) {
        int r = tile_m * BM + tid;
        s_tok[tid] = (r < cnt) ? etok[off0 + r] : -1;
    }
    __syncthreads();

    const int lane = tid & 31;
    const int wid = tid >> 5;
    const int wm = (wid & 3) * 32;
    const int wn = (wid >> 2) * 64;
    const int am = lane & 15;
    const int khalf = lane >> 4;

    v8f acc[2][4];
#pragma unroll
    for (int i = 0; i < 2; ++i)
#pragma unroll
        for (int j = 0; j < 4; ++j)
#pragma unroll
            for (int r = 0; r < 8; ++r) acc[i][j][r] = 0.f;

    const float* Wb = W1 + (size_t)e * DDIM * FDIM;
    // A loader: thread owns 16 consecutive K of one row
    const int a_row = tid >> 1;
    const int a_col = (tid & 1) * 16;
    const int atok  = s_tok[a_row];
    const float* aSrc = x + (size_t)(atok < 0 ? 0 : atok) * DDIM + a_col;
    // B loader: thread owns 16 consecutive K of one column n
    const int b_n  = tid & 127;
    const int b_kb = (tid >> 7) * 16;

    float aReg[16], bReg[16];

#define FFN1_LOAD(K0)                                                          \
    do {                                                                       \
        if (atok >= 0) {                                                       \
            _Pragma("unroll")                                                  \
            for (int c = 0; c < 16; ++c) aReg[c] = aSrc[(K0) + c];             \
        } else {                                                               \
            _Pragma("unroll")                                                  \
            for (int c = 0; c < 16; ++c) aReg[c] = 0.f;                        \
        }                                                                      \
        _Pragma("unroll")                                                      \
        for (int c = 0; c < 16; ++c)                                           \
            bReg[c] = Wb[(size_t)((K0) + b_kb + c) * FDIM + n_base + b_n];     \
    } while (0)

#define FFN1_STORE(BUF)                                                        \
    do {                                                                       \
        uint4* da = (uint4*)&lA[BUF][a_row * LDA + a_col];                     \
        uint4 va0, va1;                                                        \
        va0.x = pack2(aReg[0], aReg[1]);   va0.y = pack2(aReg[2], aReg[3]);    \
        va0.z = pack2(aReg[4], aReg[5]);   va0.w = pack2(aReg[6], aReg[7]);    \
        va1.x = pack2(aReg[8], aReg[9]);   va1.y = pack2(aReg[10], aReg[11]);  \
        va1.z = pack2(aReg[12], aReg[13]); va1.w = pack2(aReg[14], aReg[15]);  \
        da[0] = va0; da[1] = va1;                                              \
        _Pragma("unroll")                                                      \
        for (int p = 0; p < 4; ++p) {                                          \
            uint2 v;                                                           \
            v.x = pack2(bReg[4 * p], bReg[4 * p + 1]);                         \
            v.y = pack2(bReg[4 * p + 2], bReg[4 * p + 3]);                     \
            *(uint2*)&lB[BUF][b_n * LDB + b_kb + 4 * p] = v;                   \
        }                                                                      \
    } while (0)

    const int NT = DDIM / BK;
    FFN1_LOAD(0);
    FFN1_STORE(0);
    for (int it = 0; it < NT; ++it) {
        __syncthreads();
        if (it + 1 < NT) FFN1_LOAD((it + 1) * BK);
        wmma_step(lA[it & 1], lB[it & 1], acc, wm, wn, am, khalf);
        if (it + 1 < NT) FFN1_STORE((it + 1) & 1);
    }

    // epilogue: bias + exact GELU -> H bf16
#pragma unroll
    for (int sm = 0; sm < 2; ++sm) {
#pragma unroll
        for (int sn = 0; sn < 4; ++sn) {
            int col = n_base + wn + sn * 16 + am;
            float bias = b1[e * FDIM + col];
#pragma unroll
            for (int r = 0; r < 8; ++r) {
                int lrow = wm + sm * 16 + khalf * 8 + r;
                int grow = tile_m * BM + lrow;
                if (grow < cnt) {
                    float v = acc[sm][sn][r] + bias;
                    v = 0.5f * v * (1.f + erff(v * 0.70710678118654752f));
                    H[(size_t)(off0 + grow) * FDIM + col] = (bf16)v;
                }
            }
        }
    }
#undef FFN1_LOAD
#undef FFN1_STORE
}

// ---------------- FFN2: out += w * (H @ W2[e] + b2[e]) ----------------
__global__ __launch_bounds__(256) void moe_ffn2(
    const bf16* __restrict__ H, const float* __restrict__ W2,
    const float* __restrict__ b2, const int* __restrict__ etok,
    const float* __restrict__ ewt, const int* __restrict__ eoff,
    float* __restrict__ out)
{
    __shared__ bf16 lA[2][BM * LDA];
    __shared__ bf16 lB[2][BN * LDB];
    __shared__ int  s_tok[BM];

    const int e = blockIdx.y >> 5;
    const int tile_m = blockIdx.y & 31;
    const int off0 = eoff[e];
    const int cnt  = eoff[e + 1] - off0;
    if (tile_m * BM >= cnt) return;
    const int n_base = blockIdx.x * BN;

    const int tid = threadIdx.x;
    if (tid < BM) {
        int r = tile_m * BM + tid;
        s_tok[tid] = (r < cnt) ? etok[off0 + r] : -1;
    }
    __syncthreads();

    const int lane = tid & 31;
    const int wid = tid >> 5;
    const int wm = (wid & 3) * 32;
    const int wn = (wid >> 2) * 64;
    const int am = lane & 15;
    const int khalf = lane >> 4;

    v8f acc[2][4];
#pragma unroll
    for (int i = 0; i < 2; ++i)
#pragma unroll
        for (int j = 0; j < 4; ++j)
#pragma unroll
            for (int r = 0; r < 8; ++r) acc[i][j][r] = 0.f;

    const float* Wb = W2 + (size_t)e * FDIM * DDIM;
    const int b_n  = tid & 127;
    const int b_kb = (tid >> 7) * 16;

    // Async A staging: thread copies two 16B chunks of the 128x32 bf16 tile.
    // chunk id c in [0,512): row = c>>2, seg = c&3 (8 bf16 per chunk).
    const int r0 = tid >> 2,         s0 = tid & 3;
    const int r1 = (tid + 256) >> 2, s1 = (tid + 256) & 3;
    int g0 = off0 + tile_m * BM + r0; if (g0 > TOTROWS - 1) g0 = TOTROWS - 1;
    int g1 = off0 + tile_m * BM + r1; if (g1 > TOTROWS - 1) g1 = TOTROWS - 1;
    const unsigned voff0 = (unsigned)((size_t)g0 * FDIM * 2 + (size_t)s0 * 16);
    const unsigned voff1 = (unsigned)((size_t)g1 * FDIM * 2 + (size_t)s1 * 16);
    const unsigned ldsA0[2] = {
        (unsigned)(uintptr_t)&lA[0][r0 * LDA + s0 * 8],
        (unsigned)(uintptr_t)&lA[1][r0 * LDA + s0 * 8] };
    const unsigned ldsA1[2] = {
        (unsigned)(uintptr_t)&lA[0][r1 * LDA + s1 * 8],
        (unsigned)(uintptr_t)&lA[1][r1 * LDA + s1 * 8] };

    float bReg[16];

#define FFN2_ASYNC_A(BUF, K0)                                                  \
    do {                                                                       \
        const bf16* sa = H + (K0);                                             \
        asm volatile("global_load_async_to_lds_b128 %0, %1, %2"                \
                     :: "v"(ldsA0[BUF]), "v"(voff0), "s"(sa) : "memory");      \
        asm volatile("global_load_async_to_lds_b128 %0, %1, %2"                \
                     :: "v"(ldsA1[BUF]), "v"(voff1), "s"(sa) : "memory");      \
    } while (0)

#define FFN2_LOAD_B(K0)                                                        \
    do {                                                                       \
        _Pragma("unroll")                                                      \
        for (int c = 0; c < 16; ++c)                                           \
            bReg[c] = Wb[(size_t)((K0) + b_kb + c) * DDIM + n_base + b_n];     \
    } while (0)

#define FFN2_STORE_B(BUF)                                                      \
    do {                                                                       \
        _Pragma("unroll")                                                      \
        for (int p = 0; p < 4; ++p) {                                          \
            uint2 v;                                                           \
            v.x = pack2(bReg[4 * p], bReg[4 * p + 1]);                         \
            v.y = pack2(bReg[4 * p + 2], bReg[4 * p + 3]);                     \
            *(uint2*)&lB[BUF][b_n * LDB + b_kb + 4 * p] = v;                   \
        }                                                                      \
    } while (0)

    const int NT = FDIM / BK;
    FFN2_ASYNC_A(0, 0);
    FFN2_LOAD_B(0);
    FFN2_STORE_B(0);
    for (int it = 0; it < NT; ++it) {
        asm volatile("s_wait_asynccnt 0" ::: "memory");
        __syncthreads();
        if (it + 1 < NT) {
            FFN2_ASYNC_A((it + 1) & 1, (it + 1) * BK);
            FFN2_LOAD_B((it + 1) * BK);
        }
        wmma_step(lA[it & 1], lB[it & 1], acc, wm, wn, am, khalf);
        if (it + 1 < NT) FFN2_STORE_B((it + 1) & 1);
    }

    // epilogue: bias + combine weight, atomic scatter into out
#pragma unroll
    for (int sm = 0; sm < 2; ++sm) {
#pragma unroll
        for (int sn = 0; sn < 4; ++sn) {
            int col = n_base + wn + sn * 16 + am;
            float bias = b2[e * DDIM + col];
#pragma unroll
            for (int r = 0; r < 8; ++r) {
                int lrow = wm + sm * 16 + khalf * 8 + r;
                int grow = tile_m * BM + lrow;
                if (grow < cnt) {
                    int tok = s_tok[lrow];
                    float w = ewt[off0 + grow];
                    float v = (acc[sm][sn][r] + bias) * w;
                    atomicAdd(&out[(size_t)tok * DDIM + col], v);
                }
            }
        }
    }
#undef FFN2_ASYNC_A
#undef FFN2_LOAD_B
#undef FFN2_STORE_B
}

extern "C" void kernel_launch(void* const* d_in, const int* in_sizes, int n_in,
                              void* d_out, int out_size, void* d_ws, size_t ws_size,
                              hipStream_t stream) {
    const float* x  = (const float*)d_in[0];
    const float* Wr = (const float*)d_in[1];
    const float* W1 = (const float*)d_in[2];
    const float* b1 = (const float*)d_in[3];
    const float* W2 = (const float*)d_in[4];
    const float* b2 = (const float*)d_in[5];
    float* out = (float*)d_out;

    char* ws = (char*)d_ws;
    bf16* H = (bf16*)ws;
    size_t off = (size_t)TOTROWS * FDIM * sizeof(bf16);   // 67,108,864 B
    int*   topk_idx = (int*)(ws + off);   off += (size_t)NTOK * 2 * sizeof(int);
    float* topk_w   = (float*)(ws + off); off += (size_t)NTOK * 2 * sizeof(float);
    int*   etok     = (int*)(ws + off);   off += (size_t)TOTROWS * sizeof(int);
    float* ewt      = (float*)(ws + off); off += (size_t)TOTROWS * sizeof(float);
    int*   eoff     = (int*)(ws + off);   off += 256;

    moe_router<<<NTOK / 8, 256, 0, stream>>>(x, Wr, topk_idx, topk_w);
    moe_compact<<<1, 256, 0, stream>>>(topk_idx, topk_w, etok, ewt, eoff);
    moe_zero<<<(NTOK * DDIM + 255) / 256, 256, 0, stream>>>(out, NTOK * DDIM);
    dim3 g1(FDIM / BN, NEXP * 32);
    moe_ffn1<<<g1, 256, 0, stream>>>(x, W1, b1, etok, eoff, H);
    dim3 g2(DDIM / BN, NEXP * 32);
    moe_ffn2<<<g2, 256, 0, stream>>>(H, W2, b2, etok, ewt, eoff, out);
}
